// Generator_77721728188924
// MI455X (gfx1250) — matrix operand order repaired and checked by reading, not compiled, not run on hardware
//
#include <hip/hip_runtime.h>
#include <math.h>

// ---------------------------------------------------------------------------
// CDNA5 (gfx1250) WMMA types / helpers
// ---------------------------------------------------------------------------
typedef __attribute__((ext_vector_type(16))) _Float16 v16h;
typedef __attribute__((ext_vector_type(8)))  float    v8f;

union AFrag { v16h v; _Float16 h[16]; unsigned int u[8]; };
union CFrag { v8f  v; float    f[8];  };

#define WMMA_F16(A, B, C) \
  __builtin_amdgcn_wmma_f32_16x16x32_f16(false, (A), false, (B), (short)0, (C), false, false)

#define SQRT2F 1.41421356237309515f

static __device__ __forceinline__ float clampf(float v, float lo, float hi) {
  return fminf(fmaxf(v, lo), hi);
}

// load 8 consecutive f32 and convert to 8 f16 (two b128 loads, no branches)
static __device__ __forceinline__ void load8_cvt(const float* __restrict__ p, _Float16* h) {
  float4 a = *(const float4*)p;
  float4 b = *(const float4*)(p + 4);
  h[0] = (_Float16)a.x; h[1] = (_Float16)a.y; h[2] = (_Float16)a.z; h[3] = (_Float16)a.w;
  h[4] = (_Float16)b.x; h[5] = (_Float16)b.y; h[6] = (_Float16)b.z; h[7] = (_Float16)b.w;
}

// load 8 consecutive f16 (16B, one b128) into dwords u[0..3]
static __device__ __forceinline__ void load8h(const _Float16* __restrict__ p, unsigned int* u) {
  uint4 a = *(const uint4*)p;
  u[0] = a.x; u[1] = a.y; u[2] = a.z; u[3] = a.w;
}

// load 16 consecutive f16 (32B) into dwords u[0..7]
static __device__ __forceinline__ void load16h(const _Float16* __restrict__ p, unsigned int* u) {
  uint4 a = *(const uint4*)p;
  uint4 b = *(const uint4*)(p + 8);
  u[0] = a.x; u[1] = a.y; u[2] = a.z; u[3] = a.w;
  u[4] = b.x; u[5] = b.y; u[6] = b.z; u[7] = b.w;
}

// ---------------------------------------------------------------------------
// z row-normalize: x = z * rsqrt(mean(z^2) + 1e-8)     (one block per row)
// ---------------------------------------------------------------------------
__global__ void k_norm_z(const float* __restrict__ z, float* __restrict__ o) {
  __shared__ float red[256];
  int b = blockIdx.x, tid = threadIdx.x;
  float a = 0.f;
  for (int i = tid; i < 512; i += 256) { float v = z[b * 512 + i]; a += v * v; }
  red[tid] = a; __syncthreads();
  for (int s = 128; s > 0; s >>= 1) { if (tid < s) red[tid] += red[tid + s]; __syncthreads(); }
  float sc = rsqrtf(red[0] / 512.f + 1e-8f);
  for (int i = tid; i < 512; i += 256) o[b * 512 + i] = z[b * 512 + i] * sc;
}

// zero-fill a small f16 buffer
__global__ void k_zero_h(_Float16* __restrict__ p, int n) {
  int i = blockIdx.x * 256 + threadIdx.x;
  if (i < n) p[i] = (_Float16)0.f;
}

// ---------------------------------------------------------------------------
// Generic FC via WMMA: y[m,n] = gemm(x[M,K], w[N,K]^T)*wgain + bias[n]*bgain
// K multiple of 32, N multiple of 16 (grid exact). Batch rows clamped.
// ---------------------------------------------------------------------------
__global__ void k_fc_wmma(const float* __restrict__ x, const float* __restrict__ w,
                          const float* __restrict__ bias, float* __restrict__ y,
                          int M, int N, int K, int ldy,
                          float wgain, float bgain, int act) {
  const int nbase = blockIdx.x * 16, mbase = blockIdx.y * 16;
  const int lane = threadIdx.x & 31;
  const int half = lane >> 4, lm = lane & 15;
  CFrag acc; for (int j = 0; j < 8; ++j) acc.f[j] = 0.f;
  int m = mbase + lm; if (m >= M) m = M - 1;     // clamp: dup row, never stored
  const int nn = nbase + lm;                     // always < N (exact grid)
  const float* xr = x + (size_t)m * K;
  const float* wr = w + (size_t)nn * K;
  for (int kb = 0; kb < K; kb += 32) {
    AFrag af, bf;
    const float* ap = xr + kb + 8 * half;
    load8_cvt(ap, af.h);
    load8_cvt(ap + 16, af.h + 8);
    const float* bp = wr + kb + 8 * half;
    load8_cvt(bp, bf.h);
    load8_cvt(bp + 16, bf.h + 8);
    acc.v = WMMA_F16(af.v, bf.v, acc.v);
  }
#pragma unroll
  for (int j = 0; j < 8; ++j) {
    int mr = mbase + j + 8 * half;
    if (mr < M) {
      float v = acc.f[j] * wgain + bias[nn] * bgain;
      if (act) v = (v < 0.f ? 0.2f * v : v) * SQRT2F;
      y[(size_t)mr * ldy + nn] = v;
    }
  }
}

// ---------------------------------------------------------------------------
// wsq[o,i] = sum_taps w[o,i,ky,kx]^2
// ---------------------------------------------------------------------------
__global__ void k_wsq(const float* __restrict__ w, float* __restrict__ wsq, int n) {
  int i = blockIdx.x * 256 + threadIdx.x; if (i >= n) return;
  const float* p = w + (size_t)i * 9;
  float s = 0.f;
#pragma unroll
  for (int t = 0; t < 9; ++t) { float v = p[t]; s += v * v; }
  wsq[i] = s;
}

// d[b,o] = rsqrt( sum_i styles[b,i]^2 * wsq[o,i] + 1e-8 )   (styles stride 512)
__global__ void k_demodconv(const float* __restrict__ st, const float* __restrict__ wsq,
                            float* __restrict__ d, int O, int I) {
  int i = blockIdx.x * 256 + threadIdx.x; if (i >= 24 * O) return;
  int o = i % O, b = i / O;
  const float* sr = st + b * 512;
  const float* wr = wsq + (size_t)o * I;
  float s = 0.f;
  for (int k = 0; k < I; ++k) { float sv = sr[k]; s += sv * sv * wr[k]; }
  d[b * 512 + o] = rsqrtf(s + 1e-8f);
}

// d[o] = rsqrt( sum_k w[o,k]^2 + 1e-8 )
__global__ void k_demod1x1(const float* __restrict__ w, float* __restrict__ d, int O, int K) {
  int o = blockIdx.x * 256 + threadIdx.x; if (o >= O) return;
  const float* r = w + (size_t)o * K;
  float s = 0.f;
  for (int k = 0; k < K; ++k) { float v = r[k]; s += v * v; }
  d[o] = rsqrtf(s + 1e-8f);
}

// wh[o][k] (f16) = w[o][k] * d[o]   — demod folded into f16 weights (K=128)
__global__ void k_repack_1x1(const float* __restrict__ w, const float* __restrict__ dv,
                             _Float16* __restrict__ wh, int n) {
  int i = blockIdx.x * 256 + threadIdx.x; if (i >= n) return;
  wh[i] = (_Float16)(w[i] * dv[i >> 7]);
}

// ---------------------------------------------------------------------------
// Weight repack: wrep[tap][o][i] (f16) = wt[o][i][ky][kx]
// ---------------------------------------------------------------------------
__global__ void k_repack_w(const float* __restrict__ wt, _Float16* __restrict__ wrep,
                           int O, int I, int n) {
  int idx = blockIdx.x * 256 + threadIdx.x; if (idx >= n) return;   // n = 9*O*I
  int i = idx % I; int o = (idx / I) % O; int tap = idx / (I * O);
  wrep[idx] = (_Float16)wt[((size_t)o * I + i) * 9 + tap];
}

// ---------------------------------------------------------------------------
// Activation repack: xT[b][p][i] (f16) = x[b][i][p] * styles[b][i]
// Stage phase uses CDNA5 async global->LDS copies (ASYNCcnt) to land the
// 32x32 f32 tile in LDS without a VGPR round-trip; read-out phase does the
// transpose + style-multiply + f16 convert. grid: (ceil(HW/32), I/32, B).
// ---------------------------------------------------------------------------
__global__ void k_repack_x(const float* __restrict__ x, const float* __restrict__ st,
                           _Float16* __restrict__ xT, int I, int HW) {
  __shared__ float tile[32][33];
  int b = blockIdx.z;
  int pbase = blockIdx.x * 32;
  int ibase = blockIdx.y * 32;
  int tid = threadIdx.x;
  const float* sr = st + b * 512;
  int px = tid & 31, iy = tid >> 5;          // iy in 0..7
#pragma unroll
  for (int r = 0; r < 4; ++r) {
    int il = iy + r * 8;
    int p = pbase + px;
    if (p < HW) {
      unsigned ldsoff = (unsigned)(size_t)&tile[il][px];
      const float* gp = x + ((size_t)b * I + ibase + il) * HW + p;
      asm volatile("global_load_async_to_lds_b32 %0, %1, off"
                   :: "v"(ldsoff), "v"(gp) : "memory");
    } else {
      tile[il][px] = 0.f;
    }
  }
  asm volatile("s_wait_asynccnt 0" ::: "memory");
  __syncthreads();
  int ix = tid & 31, py = tid >> 5;
#pragma unroll
  for (int r = 0; r < 4; ++r) {
    int pl = py + r * 8;
    int p = pbase + pl;
    if (p < HW)
      xT[((size_t)b * HW + p) * I + ibase + ix] =
          (_Float16)(tile[ix][pl] * sr[ibase + ix]);
  }
}

// ---------------------------------------------------------------------------
// Modulated 3x3 conv as implicit GEMM (WMMA, f32 accumulate).
// Each wave: 32 out-ch x 32 pixels -> 4 accumulators, 2 A-frags x 2 B-frags
// shared (8 b128 loads / 4 WMMA per K-step). Workgroup = 4 waves = 128 M rows.
// Boundary taps resolved by pointer-select to a zero row (no inner-loop masks).
// ---------------------------------------------------------------------------
__global__ void k_modconv3x3(const _Float16* __restrict__ xT, const _Float16* __restrict__ wrep,
                             const _Float16* __restrict__ zrow,
                             const float* __restrict__ dmv, float* __restrict__ y,
                             int I, int O, int H, int W) {
  const int HW = H * W;
  const int b = blockIdx.z;
  const int wave = threadIdx.x >> 5;
  const int lane = threadIdx.x & 31;
  const int half = lane >> 4, lm = lane & 15;
  const int obase = blockIdx.y * 128 + wave * 32;     // O multiple of 128
  const int nbase = blockIdx.x * 32;
  const int p0 = nbase + lm, p1 = nbase + 16 + lm;
  const int h0 = p0 / W, w0 = p0 - h0 * W;
  const int h1 = p1 / W, w1 = p1 - h1 * W;
  const bool q0 = p0 < HW, q1 = p1 < HW;
  const int oA = obase + lm;
  const _Float16* xb = xT + (size_t)b * HW * I;
  CFrag a00, a01, a10, a11;
  for (int j = 0; j < 8; ++j) { a00.f[j] = a01.f[j] = a10.f[j] = a11.f[j] = 0.f; }

  for (int ky = 0; ky < 3; ++ky) {
    for (int kx = 0; kx < 3; ++kx) {
      const int tap = ky * 3 + kx;
      const int yy0 = h0 + ky - 1, xx0 = w0 + kx - 1;
      const int yy1 = h1 + ky - 1, xx1 = w1 + kx - 1;
      const bool in0 = q0 && (unsigned)yy0 < (unsigned)H && (unsigned)xx0 < (unsigned)W;
      const bool in1 = q1 && (unsigned)yy1 < (unsigned)H && (unsigned)xx1 < (unsigned)W;
      const _Float16* brow0 = in0 ? xb + (size_t)(yy0 * W + xx0) * I : zrow;
      const _Float16* brow1 = in1 ? xb + (size_t)(yy1 * W + xx1) * I : zrow;
      const _Float16* arow0 = wrep + (size_t)(tap * O + oA) * I;
      const _Float16* arow1 = arow0 + (size_t)16 * I;
      for (int kb = 0; kb < I; kb += 32) {
        AFrag af0, af1, b0, b1;
        const _Float16* ap0 = arow0 + kb + 16 * half;
        load8h(ap0, af0.u);
        load8h(ap0 + 8, af0.u + 4);
        const _Float16* ap1 = arow1 + kb + 16 * half;
        load8h(ap1, af1.u);
        load8h(ap1 + 8, af1.u + 4);
        __builtin_prefetch(ap0 + 32, 0, 3);   // global_prefetch_b8
        load16h(brow0 + kb + 16 * half, b0.u);
        load16h(brow1 + kb + 16 * half, b1.u);
        a00.v = WMMA_F16(af0.v, b0.v, a00.v);
        a01.v = WMMA_F16(af0.v, b1.v, a01.v);
        a10.v = WMMA_F16(af1.v, b0.v, a10.v);
        a11.v = WMMA_F16(af1.v, b1.v, a11.v);
      }
    }
  }
#pragma unroll
  for (int j = 0; j < 8; ++j) {
    int o = obase + j + 8 * half;
    float d0 = dmv[b * 512 + o];
    float d1 = dmv[b * 512 + o + 16];
    size_t base0 = ((size_t)b * O + o) * HW;
    size_t base1 = ((size_t)b * O + o + 16) * HW;
    if (q0) { y[base0 + p0] = a00.f[j] * d0; y[base1 + p0] = a10.f[j] * d1; }
    if (q1) { y[base0 + p1] = a01.f[j] * d0; y[base1 + p1] = a11.f[j] * d1; }
  }
}

// ---------------------------------------------------------------------------
// x += noise*strength; x = clamp(lrelu(x + bias)*sqrt2, +-256)   (in place)
// ---------------------------------------------------------------------------
__global__ void k_noise_act(float* __restrict__ x, const float* __restrict__ nc,
                            const float* __restrict__ ns, const float* __restrict__ bias,
                            int C, int HW, int n) {
  int i = blockIdx.x * 256 + threadIdx.x; if (i >= n) return;
  int p = i % HW; int c = (i / HW) % C;
  float v = x[i] + nc[p] * ns[0] + bias[c];
  v = (v < 0.f ? 0.2f * v : v) * SQRT2F;
  x[i] = clampf(v, -256.f, 256.f);
}

// ---------------------------------------------------------------------------
// Per-channel batch stats (training-mode BN): mean[c], rstd[c]
// ---------------------------------------------------------------------------
__global__ void k_bnstats(const float* __restrict__ x, float* __restrict__ meanv,
                          float* __restrict__ rstdv, int B, int C, int HW) {
  __shared__ float s1[256]; __shared__ float s2[256];
  int c = blockIdx.x, tid = threadIdx.x;
  float a = 0.f, q = 0.f; int tot = B * HW;
  for (int i = tid; i < tot; i += 256) {
    int b = i / HW, p = i - b * HW;
    float v = x[((size_t)b * C + c) * HW + p];
    a += v; q += v * v;
  }
  s1[tid] = a; s2[tid] = q; __syncthreads();
  for (int s = 128; s > 0; s >>= 1) {
    if (tid < s) { s1[tid] += s1[tid + s]; s2[tid] += s2[tid + s]; }
    __syncthreads();
  }
  if (tid == 0) {
    float m = s1[0] / tot;
    float var = s2[0] / tot - m * m;
    meanv[c] = m; rstdv[c] = rsqrtf(var + 1e-5f);
  }
}

// ---------------------------------------------------------------------------
// SPADE shared conv -> pixel-major f16 activation:
// actvT[b][p][o] = relu(clamp(demod 1x1 conv(seg_nearest), +-256))
// ---------------------------------------------------------------------------
__global__ void k_spade_shared(const float* __restrict__ seg, const float* __restrict__ sw,
                               const float* __restrict__ sb, _Float16* __restrict__ actvT,
                               int res, int HW, int n) {
  int i = blockIdx.x * 256 + threadIdx.x; if (i >= n) return;
  int o = i & 127; int p = (i >> 7) % HW; int b = i / (HW * 128);
  int scale = 64 / res;
  int h = p / res, w = p - h * res;
  int sy = h * scale, sx = w * scale;
  float w0 = sw[o * 2], w1 = sw[o * 2 + 1];
  float dm = rsqrtf(w0 * w0 + w1 * w1 + 1e-8f);
  float v = (w0 * seg[(((size_t)b * 2 + 0) * 64 + sy) * 64 + sx] +
             w1 * seg[(((size_t)b * 2 + 1) * 64 + sy) * 64 + sx]) * dm + sb[o];
  v = clampf(v, -256.f, 256.f);
  actvT[i] = (_Float16)fmaxf(v, 0.f);
}

// ---------------------------------------------------------------------------
// Fused SPADE apply: gamma/beta 1x1 convs via WMMA (K=128, demod pre-folded
// into f16 weights), then x = (x-mean)*rstd*(1+gamma)+beta, in place.
// 8 b128 loads / 4 WMMA per K-step per wave.
// ---------------------------------------------------------------------------
__global__ void k_spade_apply(float* __restrict__ x, const _Float16* __restrict__ actvT,
                              const _Float16* __restrict__ wgh, const float* __restrict__ bg,
                              const _Float16* __restrict__ wbh, const float* __restrict__ bb,
                              const float* __restrict__ meanv, const float* __restrict__ rstdv,
                              int C, int HW) {
  const int b = blockIdx.z;
  const int wave = threadIdx.x >> 5;
  const int lane = threadIdx.x & 31;
  const int half = lane >> 4, lm = lane & 15;
  const int obase = blockIdx.y * 64 + wave * 16;
  const int nbase = blockIdx.x * 32;
  const int p0 = nbase + lm, p1 = nbase + 16 + lm;
  const bool q0 = p0 < HW, q1 = p1 < HW;
  const int p0c = q0 ? p0 : HW - 1, p1c = q1 ? p1 : HW - 1;
  const int oA = obase + lm;                 // always < C (exact grid)
  CFrag g0, g1, e0, e1;
  for (int j = 0; j < 8; ++j) { g0.f[j] = g1.f[j] = e0.f[j] = e1.f[j] = 0.f; }
  const _Float16* ar0 = actvT + ((size_t)b * HW + p0c) * 128;
  const _Float16* ar1 = actvT + ((size_t)b * HW + p1c) * 128;
  const _Float16* wgr = wgh + (size_t)oA * 128;
  const _Float16* wbr = wbh + (size_t)oA * 128;
  for (int kb = 0; kb < 128; kb += 32) {
    AFrag agf, abf, f0, f1;
    const _Float16* gp = wgr + kb + 16 * half;
    load8h(gp, agf.u); load8h(gp + 8, agf.u + 4);
    const _Float16* bp = wbr + kb + 16 * half;
    load8h(bp, abf.u); load8h(bp + 8, abf.u + 4);
    load16h(ar0 + kb + 16 * half, f0.u);
    load16h(ar1 + kb + 16 * half, f1.u);
    g0.v = WMMA_F16(agf.v, f0.v, g0.v);
    g1.v = WMMA_F16(agf.v, f1.v, g1.v);
    e0.v = WMMA_F16(abf.v, f0.v, e0.v);
    e1.v = WMMA_F16(abf.v, f1.v, e1.v);
  }
#pragma unroll
  for (int j = 0; j < 8; ++j) {
    int o = obase + j + 8 * half;
    float mn = meanv[o], rs = rstdv[o];
    float bgo = bg[o], bbo = bb[o];
    size_t base = ((size_t)b * C + o) * HW;
    if (q0) {
      float gm = clampf(g0.f[j] + bgo, -256.f, 256.f);
      float bt = clampf(e0.f[j] + bbo, -256.f, 256.f);
      float xv = x[base + p0];
      x[base + p0] = (xv - mn) * rs * (1.f + gm) + bt;
    }
    if (q1) {
      float gm = clampf(g1.f[j] + bgo, -256.f, 256.f);
      float bt = clampf(e1.f[j] + bbo, -256.f, 256.f);
      float xv = x[base + p1];
      x[base + p1] = (xv - mn) * rs * (1.f + gm) + bt;
    }
  }
}

// ---------------------------------------------------------------------------
// toRGB: y[b,o,p] = clamp(sum_c x[b,c,p]*styles[b,c]*w[o,c] + bias[o], +-256)
// ---------------------------------------------------------------------------
__global__ void k_torgb(const float* __restrict__ x, const float* __restrict__ w,
                        const float* __restrict__ st, const float* __restrict__ bias,
                        float* __restrict__ y, int C, int HW) {
  int b = blockIdx.y;
  int p = blockIdx.x * 256 + threadIdx.x; if (p >= HW) return;
  const float* sr = st + b * 512;
  float a0 = 0.f, a1 = 0.f, a2 = 0.f;
  for (int c = 0; c < C; ++c) {
    float xv = x[((size_t)b * C + c) * HW + p] * sr[c];
    a0 += w[c] * xv; a1 += w[C + c] * xv; a2 += w[2 * C + c] * xv;
  }
  size_t base = (size_t)b * 3 * HW;
  y[base + p]          = clampf(a0 + bias[0], -256.f, 256.f);
  y[base + HW + p]     = clampf(a1 + bias[1], -256.f, 256.f);
  y[base + 2 * HW + p] = clampf(a2 + bias[2], -256.f, 256.f);
}

// ---------------------------------------------------------------------------
// upfirdn2d(up=2, pad=(2,1), [1,3,3,1] outer /9) with optional additive input
// ---------------------------------------------------------------------------
__global__ void k_upsample_add(const float* __restrict__ x, const float* __restrict__ add,
                               float* __restrict__ out, int C, int Hin, int n) {
  int i = blockIdx.x * 256 + threadIdx.x; if (i >= n) return;
  int Hout = Hin * 2;
  int ox = i % Hout; int t2 = i / Hout;
  int oy = t2 % Hout; int t3 = t2 / Hout;
  int c = t3 % C; int b = t3 / C;
  const float f1[4] = {1.f, 3.f, 3.f, 1.f};
  const float* xs = x + ((size_t)b * C + c) * Hin * Hin;
  float acc = 0.f;
#pragma unroll
  for (int ty = 0; ty < 4; ++ty) {
    int ey = oy - 2 + ty; if (ey & 1) continue;
    int iy = ey >> 1; if ((unsigned)iy >= (unsigned)Hin) continue;
#pragma unroll
    for (int tx = 0; tx < 4; ++tx) {
      int ex = ox - 2 + tx; if (ex & 1) continue;
      int ix = ex >> 1; if ((unsigned)ix >= (unsigned)Hin) continue;
      acc += f1[ty] * f1[tx] * (1.f / 9.f) * xs[iy * Hin + ix];
    }
  }
  if (add) acc += add[i];
  out[i] = acc;
}

// Tile learned constant: x[b,c,p] = const[b%6, c, p]   (512ch, 4x4)
__global__ void k_tile_const(const float* __restrict__ c, float* __restrict__ o, int n) {
  int i = blockIdx.x * 256 + threadIdx.x; if (i >= n) return;
  int p = i & 15; int ch = (i >> 4) & 511; int b = i >> 13;
  o[i] = c[(((size_t)(b % 6) << 9) + ch) * 16 + p];
}

// ---------------------------------------------------------------------------
// Host orchestration
// ---------------------------------------------------------------------------
extern "C" void kernel_launch(void* const* d_in, const int* in_sizes, int n_in,
                              void* d_out, int out_size, void* d_ws, size_t ws_size,
                              hipStream_t stream) {
  (void)in_sizes; (void)n_in; (void)out_size; (void)ws_size;
  int t = 0;
  auto nx = [&]() -> const float* { return (const float*)d_in[t++]; };

  const float* z   = nx();
  const float* sil = nx();
  const float *mw[8], *mb[8];
  for (int i = 0; i < 8; ++i) { mw[i] = nx(); mb[i] = nx(); }

  struct ConvP  { const float *aw, *ab, *w, *nc, *ns, *b; };
  struct SpadeP { const float *sw, *sb, *gw, *gb, *bw, *bb; };
  struct RgbP   { const float *aw, *ab, *w, *b; };
  auto rdC = [&]() { ConvP p;  p.aw = nx(); p.ab = nx(); p.w = nx(); p.nc = nx(); p.ns = nx(); p.b = nx(); return p; };
  auto rdS = [&]() { SpadeP p; p.sw = nx(); p.sb = nx(); p.gw = nx(); p.gb = nx(); p.bw = nx(); p.bb = nx(); return p; };
  auto rdR = [&]() { RgbP p;   p.aw = nx(); p.ab = nx(); p.w = nx(); p.b = nx(); return p; };

  const float* procst = nx();
  ConvP pc1 = rdC(); SpadeP ps1 = rdS(); RgbP pr = rdR();
  ConvP c0[4], c1[4]; SpadeP s0[4], s1[4]; RgbP rg[4];
  for (int i = 0; i < 4; ++i) { c0[i] = rdC(); c1[i] = rdC(); s0[i] = rdS(); s1[i] = rdS(); rg[i] = rdR(); }

  // ---- workspace carve (256B aligned chunks) ----
  char* wsb = (char*)d_ws;
  size_t off = 0;
  auto alloc = [&](size_t bytes) -> void* {
    size_t r = (off + 255) & ~(size_t)255; off = r + bytes; return (void*)(wsb + r);
  };
  float* wvA    = (float*)alloc((size_t)24 * 512 * 4);
  float* wvB    = (float*)alloc((size_t)24 * 512 * 4);
  float* styles = (float*)alloc((size_t)24 * 512 * 4);
  float* demod  = (float*)alloc((size_t)24 * 512 * 4);
  float* wsq    = (float*)alloc((size_t)512 * 512 * 4);
  float* dg     = (float*)alloc(512 * 4);
  float* dbv    = (float*)alloc(512 * 4);
  float* meanB  = (float*)alloc(512 * 4);
  float* rstdB  = (float*)alloc(512 * 4);
  float* ybuf   = (float*)alloc((size_t)24 * 3 * 64 * 64 * 4);
  float* imA    = (float*)alloc((size_t)24 * 3 * 64 * 64 * 4);
  float* imB    = (float*)alloc((size_t)24 * 3 * 64 * 64 * 4);
  float* XA     = (float*)alloc((size_t)24 * 256 * 64 * 64 * 4);
  float* XB     = (float*)alloc((size_t)24 * 256 * 64 * 64 * 4);
  _Float16* xT    = (_Float16*)alloc((size_t)24 * 64 * 64 * 256 * 2); // [b][HW][I] max
  _Float16* wrep  = (_Float16*)alloc((size_t)9 * 512 * 512 * 2);      // [tap][O][I] max
  _Float16* actvT = (_Float16*)alloc((size_t)24 * 64 * 64 * 128 * 2); // [b][HW][128] max
  _Float16* wgh   = (_Float16*)alloc((size_t)512 * 128 * 2);          // f16 gamma weights
  _Float16* wbh   = (_Float16*)alloc((size_t)512 * 128 * 2);          // f16 beta weights
  _Float16* zrow  = (_Float16*)alloc((size_t)512 * 2);                // zero row for halo

  const float rs512 = 1.f / sqrtf(512.f);

  k_zero_h<<<2, 256, 0, stream>>>(zrow, 512);

  // ---- mapping network ----
  k_norm_z<<<24, 256, 0, stream>>>(z, wvA);
  float* src = wvA; float* dst = wvB;
  const float wgm = 0.01f * rs512;
  for (int i = 0; i < 8; ++i) {
    dim3 g(512 / 16, 2);
    k_fc_wmma<<<g, 32, 0, stream>>>(src, mw[i], mb[i], dst, 24, 512, 512, 512, wgm, 0.01f, 1);
    float* tp = src; src = dst; dst = tp;
  }
  float* wvec = src;   // [24,512]; all 12 ws are identical

  // ---- stage helpers ----
  auto runConv = [&](const float* xin, float* xout, const ConvP& p, int I, int O, int H) {
    dim3 gf(I / 16, 2);
    k_fc_wmma<<<gf, 32, 0, stream>>>(wvec, p.aw, p.ab, styles, 24, I, 512, 512, rs512, 1.f, 0);
    k_wsq<<<(O * I + 255) / 256, 256, 0, stream>>>(p.w, wsq, O * I);
    k_demodconv<<<(24 * O + 255) / 256, 256, 0, stream>>>(styles, wsq, demod, O, I);
    { int n = 9 * O * I;
      k_repack_w<<<(n + 255) / 256, 256, 0, stream>>>(p.w, wrep, O, I, n); }
    { dim3 gr((H * H + 31) / 32, I / 32, 24);
      k_repack_x<<<gr, 256, 0, stream>>>(xin, styles, xT, I, H * H); }
    dim3 gc((H * H + 31) / 32, O / 128, 24);
    k_modconv3x3<<<gc, 128, 0, stream>>>(xT, wrep, zrow, demod, xout, I, O, H, H);
  };
  auto runNoise = [&](float* x, const ConvP& p, int C, int res) {
    int n = 24 * C * res * res;
    k_noise_act<<<(n + 255) / 256, 256, 0, stream>>>(x, p.nc, p.ns, p.b, C, res * res, n);
  };
  auto runSpade = [&](float* x, const SpadeP& s, int C, int res) {
    int HW = res * res;
    k_bnstats<<<C, 256, 0, stream>>>(x, meanB, rstdB, 24, C, HW);
    int na = 24 * 128 * HW;
    k_spade_shared<<<(na + 255) / 256, 256, 0, stream>>>(sil, s.sw, s.sb, actvT, res, HW, na);
    k_demod1x1<<<(C + 255) / 256, 256, 0, stream>>>(s.gw, dg, C, 128);
    k_demod1x1<<<(C + 255) / 256, 256, 0, stream>>>(s.bw, dbv, C, 128);
    int n1 = C * 128;
    k_repack_1x1<<<(n1 + 255) / 256, 256, 0, stream>>>(s.gw, dg, wgh, n1);
    k_repack_1x1<<<(n1 + 255) / 256, 256, 0, stream>>>(s.bw, dbv, wbh, n1);
    dim3 g((HW + 31) / 32, C / 64, 24);
    k_spade_apply<<<g, 128, 0, stream>>>(x, actvT, wgh, s.gb, wbh, s.bb, meanB, rstdB, C, HW);
  };
  auto runRGB = [&](const float* x, const RgbP& r, float* y, int C, int res) {
    float sc = 1.f / sqrtf((float)C);
    dim3 gf(C / 16, 2);
    k_fc_wmma<<<gf, 32, 0, stream>>>(wvec, r.aw, r.ab, styles, 24, C, 512, 512, rs512 * sc, sc, 0);
    dim3 g((res * res + 255) / 256, 24);
    k_torgb<<<g, 256, 0, stream>>>(x, r.w, styles, r.b, y, C, res * res);
  };
  auto runUp = [&](const float* x, const float* add, float* out, int C, int Hin) {
    int n = 24 * C * Hin * Hin * 4;
    k_upsample_add<<<(n + 255) / 256, 256, 0, stream>>>(x, add, out, C, Hin, n);
  };

  // ---- prologue (res 4) ----
  { int n = 24 * 512 * 16; k_tile_const<<<(n + 255) / 256, 256, 0, stream>>>(procst, XA, n); }
  runConv(XA, XB, pc1, 512, 512, 4);
  runNoise(XB, pc1, 512, 4);
  runSpade(XB, ps1, 512, 4);
  runRGB(XB, pr, imA, 512, 4);

  float* xCur = XB; float* xAlt = XA;
  float* imCur = imA;
  const int inC[4]  = {512, 512, 512, 512};
  const int outC[4] = {512, 512, 512, 256};
  const int resv[4] = {8, 16, 32, 64};

  for (int i = 0; i < 4; ++i) {
    int r = resv[i], hr = r / 2;
    // conv0 at input resolution, then smooth 2x upsample
    runConv(xCur, xAlt, c0[i], inC[i], outC[i], hr);
    runUp(xAlt, nullptr, xCur, outC[i], hr);
    runNoise(xCur, c0[i], outC[i], r);
    runSpade(xCur, s0[i], outC[i], r);
    // conv1 at output resolution
    runConv(xCur, xAlt, c1[i], outC[i], outC[i], r);
    runNoise(xAlt, c1[i], outC[i], r);
    runSpade(xAlt, s1[i], outC[i], r);
    { float* tp = xCur; xCur = xAlt; xAlt = tp; }
    // skip connection: img = up(img) + toRGB(x)
    runRGB(xCur, rg[i], ybuf, outC[i], r);
    float* imNext = (i == 3) ? (float*)d_out : (imCur == imA ? imB : imA);
    runUp(imCur, ybuf, imNext, 3, hr);
    imCur = imNext;
  }
}
